// MultiHeadAttention_70179765617325
// MI455X (gfx1250) — matrix operand order repaired and checked by reading, not compiled, not run on hardware
//
#include <hip/hip_runtime.h>

// ---------------------------------------------------------------------------
// MultiHeadAttention forward for MI455X (gfx1250), wave32 + WMMA bf16.
// B=4, S=2048, D=1024, H=16, d=64.
// Pipeline: cvt(x)->bf16, transpose-cvt(W*)->bf16, mask->f32 bias;
// double-buffered WMMA GEMMs staged with global_load_async_to_lds_b128;
// flash attention (64-key chunks, online softmax, rowsum via WMMA-with-ones).
// ---------------------------------------------------------------------------

typedef __attribute__((ext_vector_type(16))) __bf16       v16bf;
typedef __attribute__((ext_vector_type(8)))  float        v8f;
typedef __attribute__((ext_vector_type(4)))  unsigned int u32x4;

#define BATCH  4
#define SEQ    2048
#define DMODEL 1024
#define NHEAD  16
#define DHEAD  64
#define LDK    72   // GEMM LDS row stride (64 + 8 pad), 144B rows, 16B aligned

union FragU { v16bf v; u32x4 q[2]; __bf16 h[16]; };

__device__ inline v8f v8f_zero() {
  v8f z;
#pragma unroll
  for (int i = 0; i < 8; ++i) z[i] = 0.0f;
  return z;
}

__device__ inline v8f wmma_bf16(v16bf a, v16bf b, v8f c) {
  return __builtin_amdgcn_wmma_f32_16x16x32_bf16(
      false, a, false, b, (short)0, c, false, false);
}

// A-fragment (16x32 MxK): lane<16 -> row M=lane, elems {K0..7,K16..23};
// lane>=16 -> row M=lane-16, elems {K8..15,K24..31}.
__device__ inline v16bf load_a_frag(const __bf16* src, int ld, int lane) {
  const int half = lane >> 4, m = lane & 15;
  const __bf16* p = src + m * ld + 8 * half;
  FragU f;
  f.q[0] = *(const u32x4*)(p);
  f.q[1] = *(const u32x4*)(p + 16);
  return f.v;
}

// B-fragment (32x16 KxN) from K-major array bt[n*ld + k]:
// lane -> col N=lane%16, elems {K 8*(lane/16)+0..7, K 16+8*(lane/16)+0..7}.
__device__ inline v16bf load_b_frag(const __bf16* bt, int ld, int lane) {
  const int half = lane >> 4, n = lane & 15;
  const __bf16* p = bt + n * ld + 8 * half;
  FragU f;
  f.q[0] = *(const u32x4*)(p);
  f.q[1] = *(const u32x4*)(p + 16);
  return f.v;
}

// Reduce max across the 16 lanes holding one C-matrix row.
__device__ inline float rowmax16(float v) {
#pragma unroll
  for (int m = 1; m <= 8; m <<= 1) v = fmaxf(v, __shfl_xor(v, m, 32));
  return v;
}

// --- CDNA5 async copy: 16B global -> LDS, tracked by ASYNCcnt --------------
__device__ inline void async_ld128(unsigned lds_off, const void* gaddr) {
  asm volatile("global_load_async_to_lds_b128 %0, %1, off"
               :: "v"(lds_off), "v"(gaddr) : "memory");
}
__device__ inline void wait_async0() {
  asm volatile("s_wait_asynccnt 0x0" ::: "memory");
}
__device__ inline unsigned lds_off32(const void* p) {
  return (unsigned)(size_t)p;   // flat LDS addr: low 32 bits are the offset
}

// ---------------------------------------------------------------------------
// Prep 1: fp32 -> bf16 elementwise (for x).
// ---------------------------------------------------------------------------
__global__ __launch_bounds__(256) void cvt_bf16_kernel(
    const float* __restrict__ s, __bf16* __restrict__ d, int nvec) {
  int i = blockIdx.x * 256 + threadIdx.x;
  if (i >= nvec) return;
  float4 a = ((const float4*)s)[2 * (size_t)i];
  float4 b = ((const float4*)s)[2 * (size_t)i + 1];
  union { __bf16 h[8]; u32x4 q; } u;
  u.h[0] = (__bf16)a.x; u.h[1] = (__bf16)a.y; u.h[2] = (__bf16)a.z; u.h[3] = (__bf16)a.w;
  u.h[4] = (__bf16)b.x; u.h[5] = (__bf16)b.y; u.h[6] = (__bf16)b.z; u.h[7] = (__bf16)b.w;
  ((u32x4*)d)[i] = u.q;
}

// ---------------------------------------------------------------------------
// Prep 2: WT[n][k] = bf16(W[k][n])  (1024x1024, 32x32 LDS-tiled transpose)
// ---------------------------------------------------------------------------
__global__ __launch_bounds__(256) void transpose_cvt_kernel(
    const float* __restrict__ W, __bf16* __restrict__ WT) {
  __shared__ float tile[32][33];
  const int tx = threadIdx.x, ty = threadIdx.y;
  const int bx = blockIdx.x * 32, by = blockIdx.y * 32;
#pragma unroll
  for (int i = ty; i < 32; i += 8)
    tile[i][tx] = W[(size_t)(by + i) * DMODEL + bx + tx];
  __syncthreads();
#pragma unroll
  for (int i = ty; i < 32; i += 8)
    WT[(size_t)(bx + i) * DMODEL + by + tx] = (__bf16)tile[tx][i];
}

// ---------------------------------------------------------------------------
// Prep 3: key mask -> additive fp32 bias (0 or NEG_INF).
// ---------------------------------------------------------------------------
__global__ __launch_bounds__(256) void mask_bias_kernel(
    const unsigned char* __restrict__ m, float* __restrict__ bias, int n) {
  int i = blockIdx.x * 256 + threadIdx.x;
  if (i < n) bias[i] = m[i] ? 0.0f : -1e30f;
}

// ---------------------------------------------------------------------------
// WMMA GEMM: C(128x128 per block) = A(bf16, MxK row-major) @ WT^T
//   A: (B*S) x 1024 bf16,  WT: N-major 1024x1024 bf16 (WT[n][k] = W[k][n]).
// K-step 64, double-buffered LDS filled by async-to-LDS b128.
// MODE 0: out bf16 head-split (b,h,s,d)     [Q, K]
// MODE 1: out bf16 transposed  (b,h,d,s)    [V]
// MODE 2: out fp32 row-major   (B*S, 1024)  [final projection]
// ---------------------------------------------------------------------------
template <int MODE>
__global__ __launch_bounds__(256) void gemm_bf16_kernel(
    const __bf16* __restrict__ A, const __bf16* __restrict__ WT,
    void* __restrict__ outv, float alpha) {
  __shared__ __align__(16) __bf16 As[2][128 * LDK];
  __shared__ __align__(16) __bf16 Bs[2][128 * LDK];

  const int tid  = threadIdx.x;
  const int lane = tid & 31, w = tid >> 5;
  const int wr = w & 3, wc = w >> 2;          // 4x2 wave grid -> 32x64 / wave
  const int m0 = blockIdx.y * 128, n0 = blockIdx.x * 128;

  v8f acc[2][4];
#pragma unroll
  for (int i = 0; i < 2; ++i)
#pragma unroll
    for (int j = 0; j < 4; ++j) acc[i][j] = v8f_zero();

  auto stage = [&](int buf, int k0) {
#pragma unroll
    for (int i = 0; i < 4; ++i) {
      int idx = tid + 256 * i;                 // 1024 chunks of 16B per tile
      int row = idx >> 3, c = idx & 7;         // 128 rows x 8 chunks
      async_ld128(lds_off32(&As[buf][row * LDK + c * 8]),
                  &A[(size_t)(m0 + row) * DMODEL + k0 + c * 8]);
      async_ld128(lds_off32(&Bs[buf][row * LDK + c * 8]),
                  &WT[(size_t)(n0 + row) * DMODEL + k0 + c * 8]);
    }
  };

  stage(0, 0);
  wait_async0();
  __syncthreads();

  for (int kt = 0; kt < DMODEL / 64; ++kt) {
    const int buf = kt & 1;
    if (kt + 1 < DMODEL / 64) stage(buf ^ 1, (kt + 1) * 64);  // prefetch next

#pragma unroll
    for (int kc = 0; kc < 2; ++kc) {
      v16bf a0 = load_a_frag(&As[buf][(wr * 32 + 0)  * LDK + kc * 32], LDK, lane);
      v16bf a1 = load_a_frag(&As[buf][(wr * 32 + 16) * LDK + kc * 32], LDK, lane);
#pragma unroll
      for (int nt = 0; nt < 4; ++nt) {
        v16bf b = load_b_frag(&Bs[buf][(wc * 64 + nt * 16) * LDK + kc * 32], LDK, lane);
        acc[0][nt] = wmma_bf16(a0, b, acc[0][nt]);
        acc[1][nt] = wmma_bf16(a1, b, acc[1][nt]);
      }
    }
    wait_async0();        // next-tile asyncs landed
    __syncthreads();      // all waves done reading current buffer
  }

  const int half = lane >> 4, ln = lane & 15;
#pragma unroll
  for (int mt = 0; mt < 2; ++mt)
#pragma unroll
    for (int nt = 0; nt < 4; ++nt)
#pragma unroll
      for (int r = 0; r < 8; ++r) {
        int m = m0 + wr * 32 + mt * 16 + r + 8 * half;
        int n = n0 + wc * 64 + nt * 16 + ln;
        float val = acc[mt][nt][r] * alpha;
        if (MODE == 2) {
          ((float*)outv)[(size_t)m * DMODEL + n] = val;
        } else {
          int b = m >> 11, s = m & (SEQ - 1);
          int h = n >> 6,  j = n & (DHEAD - 1);
          size_t addr = (MODE == 1)
              ? ((size_t)(b * NHEAD + h) * DHEAD + j) * SEQ + s
              : ((size_t)(b * NHEAD + h) * SEQ + s) * DHEAD + j;
          ((__bf16*)outv)[addr] = (__bf16)val;
        }
      }
}

// ---------------------------------------------------------------------------
// Flash attention: block = (b*h, 128 q rows); wave owns 16 q rows; keys
// streamed in 64-key chunks with online softmax.  Row max via wave32 shfl
// (ds_bpermute); row sum via WMMA against an all-ones B matrix (free layout
// match with the C fragment).  Mask folded in as the C-input of the QK WMMA.
// Q,K: (B,H,S,d) bf16 (Q pre-scaled by d^-0.5); V: (B,H,d,S) bf16.
// ---------------------------------------------------------------------------
__global__ __launch_bounds__(256) void flash_attn_kernel(
    const __bf16* __restrict__ Q, const __bf16* __restrict__ Kmat,
    const __bf16* __restrict__ VT, const float* __restrict__ bias,
    __bf16* __restrict__ attn) {
  __shared__ __align__(16) __bf16 Plds[8][16 * 64];   // per-wave 16x64 P tile

  const int tid  = threadIdx.x;
  const int lane = tid & 31, w = tid >> 5;
  const int half = lane >> 4, ln = lane & 15;
  const int bh = blockIdx.y;
  const int b  = bh >> 4, h = bh & (NHEAD - 1);
  const int qbase = blockIdx.x * 128 + w * 16;

  const __bf16* qptr   = Q    + ((size_t)bh * SEQ + qbase) * DHEAD;
  const __bf16* kbase  = Kmat + (size_t)bh * SEQ * DHEAD;
  const __bf16* vtbase = VT   + (size_t)bh * DHEAD * SEQ;
  const float*  bptr   = bias + (size_t)b * SEQ;

  v16bf qa0 = load_a_frag(qptr + 0,  DHEAD, lane);
  v16bf qa1 = load_a_frag(qptr + 32, DHEAD, lane);

  FragU onesf;
#pragma unroll
  for (int i = 0; i < 16; ++i) onesf.h[i] = (__bf16)1.0f;
  const v16bf ones = onesf.v;

  float mrow[8], lrow[8];
  v8f o[4];
#pragma unroll
  for (int r = 0; r < 8; ++r) { mrow[r] = -1e30f; lrow[r] = 0.0f; }
#pragma unroll
  for (int nt = 0; nt < 4; ++nt) o[nt] = v8f_zero();

  __bf16* pl = &Plds[w][0];

  for (int kt = 0; kt < SEQ; kt += 64) {
    // pull next K chunk toward the caches (per-lane 128B spread covers 4KB)
    if (kt + 64 < SEQ)
      __builtin_prefetch((const char*)(kbase + (size_t)(kt + 64) * DHEAD) +
                             lane * 128, 0, 1);

    // ---- logits: four 16-key tiles; mask bias rides in as the C input ----
    v8f cl[4];
#pragma unroll
    for (int t = 0; t < 4; ++t) {
      v16bf kb0 = load_b_frag(kbase + (size_t)(kt + 16 * t) * DHEAD + 0,  DHEAD, lane);
      v16bf kb1 = load_b_frag(kbase + (size_t)(kt + 16 * t) * DHEAD + 32, DHEAD, lane);
      float mb = bptr[kt + 16 * t + ln];
      v8f c;
#pragma unroll
      for (int r = 0; r < 8; ++r) c[r] = mb;
      c = wmma_bf16(qa0, kb0, c);
      c = wmma_bf16(qa1, kb1, c);
      cl[t] = c;
    }

    // ---- online softmax: new max + exponentials ----
    float scale[8];
#pragma unroll
    for (int r = 0; r < 8; ++r) {
      float tm = fmaxf(fmaxf(cl[0][r], cl[1][r]), fmaxf(cl[2][r], cl[3][r]));
      tm = rowmax16(tm);
      float nm = fmaxf(mrow[r], tm);
#pragma unroll
      for (int t = 0; t < 4; ++t) cl[t][r] = __expf(cl[t][r] - nm);
      scale[r] = __expf(mrow[r] - nm);
      mrow[r]  = nm;
    }

    // ---- relayout P (C-layout) -> A-fragments via per-wave LDS ----
#pragma unroll
    for (int r = 0; r < 8; ++r) {
      int M = r + 8 * half;
#pragma unroll
      for (int t = 0; t < 4; ++t)
        pl[M * 64 + 16 * t + ln] = (__bf16)cl[t][r];
    }
    asm volatile("s_wait_dscnt 0x0" ::: "memory");
    v16bf pa0 = load_a_frag(pl + 0,  64, lane);
    v16bf pa1 = load_a_frag(pl + 32, 64, lane);

    // ---- row sums of P via WMMA against ones (replaces shfl reduction) ----
    v8f rsum = wmma_bf16(pa0, ones, v8f_zero());
    rsum     = wmma_bf16(pa1, ones, rsum);
#pragma unroll
    for (int r = 0; r < 8; ++r)
      lrow[r] = lrow[r] * scale[r] + rsum[r];

    // ---- rescale O, then O += P @ V ----
#pragma unroll
    for (int nt = 0; nt < 4; ++nt)
#pragma unroll
      for (int r = 0; r < 8; ++r) o[nt][r] *= scale[r];
#pragma unroll
    for (int nt = 0; nt < 4; ++nt) {
      v16bf vb0 = load_b_frag(vtbase + (size_t)(nt * 16) * SEQ + kt,      SEQ, lane);
      v16bf vb1 = load_b_frag(vtbase + (size_t)(nt * 16) * SEQ + kt + 32, SEQ, lane);
      o[nt] = wmma_bf16(pa0, vb0, o[nt]);
      o[nt] = wmma_bf16(pa1, vb1, o[nt]);
    }
  }

#pragma unroll
  for (int r = 0; r < 8; ++r) lrow[r] = 1.0f / fmaxf(lrow[r], 1e-30f);
#pragma unroll
  for (int nt = 0; nt < 4; ++nt)
#pragma unroll
    for (int r = 0; r < 8; ++r) {
      int s   = qbase + r + 8 * half;
      int col = h * DHEAD + nt * 16 + ln;
      attn[((size_t)b * SEQ + s) * DMODEL + col] = (__bf16)(o[nt][r] * lrow[r]);
    }
}

// ---------------------------------------------------------------------------
extern "C" void kernel_launch(void* const* d_in, const int* in_sizes, int n_in,
                              void* d_out, int out_size, void* d_ws, size_t ws_size,
                              hipStream_t stream) {
  (void)in_sizes; (void)n_in; (void)out_size; (void)ws_size;
  const float* x  = (const float*)d_in[0];
  const unsigned char* mask = (const unsigned char*)d_in[1];
  const float* Wq = (const float*)d_in[2];
  const float* Wk = (const float*)d_in[3];
  const float* Wv = (const float*)d_in[4];
  const float* Wo = (const float*)d_in[5];
  float* out = (float*)d_out;

  const size_t BSD = (size_t)BATCH * SEQ * DMODEL;   // 8M elems
  const size_t DD  = (size_t)DMODEL * DMODEL;        // 1M elems
  __bf16* q    = (__bf16*)d_ws;
  __bf16* k    = q    + BSD;
  __bf16* v    = k    + BSD;       // stored transposed (B,H,d,S)
  __bf16* attn = v    + BSD;       // (B,S,D)
  __bf16* xb   = attn + BSD;       // x in bf16
  __bf16* wtq  = xb   + BSD;       // W^T bf16
  __bf16* wtk  = wtq  + DD;
  __bf16* wtv  = wtk  + DD;
  __bf16* wto  = wtv  + DD;
  float*  bias = (float*)(wto + DD);  // (B,S) additive mask bias

  dim3 blk(256);

  // prep: conversions / transposes / mask bias (trivial vs. GEMM FLOPs)
  cvt_bf16_kernel<<<(int)(BSD / 8 / 256), blk, 0, stream>>>(x, xb, (int)(BSD / 8));
  dim3 tb(32, 8), tg(DMODEL / 32, DMODEL / 32);
  transpose_cvt_kernel<<<tg, tb, 0, stream>>>(Wq, wtq);
  transpose_cvt_kernel<<<tg, tb, 0, stream>>>(Wk, wtk);
  transpose_cvt_kernel<<<tg, tb, 0, stream>>>(Wv, wtv);
  transpose_cvt_kernel<<<tg, tb, 0, stream>>>(Wo, wto);
  mask_bias_kernel<<<(BATCH * SEQ) / 256, blk, 0, stream>>>(
      mask, bias, BATCH * SEQ);

  dim3 gg(DMODEL / 128, (BATCH * SEQ) / 128);        // 8 x 64
  gemm_bf16_kernel<0><<<gg, blk, 0, stream>>>(xb, wtq, q, 0.125f);  // d^-0.5
  gemm_bf16_kernel<0><<<gg, blk, 0, stream>>>(xb, wtk, k, 1.0f);
  gemm_bf16_kernel<1><<<gg, blk, 0, stream>>>(xb, wtv, v, 1.0f);

  dim3 gf(SEQ / 128, BATCH * NHEAD);                 // 16 x 64
  flash_attn_kernel<<<gf, blk, 0, stream>>>(q, k, v, bias, attn);

  gemm_bf16_kernel<2><<<gg, blk, 0, stream>>>(attn, wto, out, 1.0f);
}
